// Net_29815662969021
// MI455X (gfx1250) — matrix operand order repaired and checked by reading, not compiled, not run on hardware
//
#include <hip/hip_runtime.h>
#include <hip/hip_bf16.h>
#include <stdint.h>

typedef __attribute__((ext_vector_type(16))) _Float16 v16h;
typedef __attribute__((ext_vector_type(8)))  _Float16 v8h;
typedef __attribute__((ext_vector_type(8)))  float    v8f;

#define Bn   4
#define GPn  6
#define Gn   24          // Bn*GPn
#define Nn   1024
#define DINn 128
#define Dn   128
#define En   16384

// ---------------- workspace layout (bytes) ----------------
static constexpr size_t OFF_X16  = 0;                         // [G,N,128] f16      6,291,456
static constexpr size_t OFF_W16  = 6291456;                   // gcn_W^T f16          131,072
static constexpr size_t OFF_WS16 = 6422528;                   // Ws^T f16             131,072
static constexpr size_t OFF_DEG  = 6553600;                   // [G,4,N] f32          393,216
static constexpr size_t OFF_XW   = 6946816;                   // [G,N,128] f32 (h f16 aliases)
static constexpr size_t OFF_ACC  = 19529728;                  // [G,N,128] f32 (xn f16 aliases)
static constexpr size_t OFF_XTMP = 32112640;                  // [G,N,128] f32
static constexpr size_t OFF_GMAT = 44695552;                  // [B,768] f32
static constexpr size_t OFF_SIM  = 44707840;                  // [B,6,6] f32 accumulators

// ---------------- WMMA operand loaders (ISA 7.12.2 layouts, wave32) ----------------
// A 16xK row-major: lane(half=lane>>4,m=lane&15) holds two contiguous 8-half chunks:
//   K = k0+8*half+0..7  and  K = k0+16+8*half+0..7
// B as Bt (NxK row-major, representing B=Bt^T): lane holds two contiguous 8-half chunks at
//   K = k0+8*half+0..7  and  K = k0+16+8*half+0..7  of row n=lane&15
// C f32 16x16: c[r] -> row (r+8*half), col (lane&15)
__device__ __forceinline__ v16h pack16(v8h lo, v8h hi) {
  return __builtin_shufflevector(lo, hi, 0, 1, 2, 3, 4, 5, 6, 7,
                                 8, 9, 10, 11, 12, 13, 14, 15);
}

__device__ __forceinline__ v16h loadA(const _Float16* __restrict__ A, int lda,
                                      int lane, int k0) {
  const int half = lane >> 4, m = lane & 15;
  const _Float16* p = A + m * lda + k0 + 8 * half;
  return pack16(*(const v8h*)p, *(const v8h*)(p + 16));
}

__device__ __forceinline__ v16h loadBt(const _Float16* __restrict__ Bt, int ldbt,
                                       int lane, int k0) {
  const int half = lane >> 4, n = lane & 15;
  const _Float16* p = Bt + n * ldbt + k0 + 8 * half;
  return pack16(*(const v8h*)p, *(const v8h*)(p + 16));
}

#define WMMA_F16(a, b, c) \
  __builtin_amdgcn_wmma_f32_16x16x32_f16(false, (a), false, (b), (short)0, (c), false, false)

// ---------------- simple elementwise kernels ----------------
__global__ void k_convert_x(const float* __restrict__ x, _Float16* __restrict__ x16, int n) {
  int i = blockIdx.x * blockDim.x + threadIdx.x;
  if (i < n) x16[i] = (_Float16)x[i];
}

// transpose-convert both weight tensors: wt[c][n][k] = W[c][k][n]
__global__ void k_convert_w(const float* __restrict__ gw, const float* __restrict__ wsm,
                            _Float16* __restrict__ gwt16, _Float16* __restrict__ wst16) {
  int i = blockIdx.x * blockDim.x + threadIdx.x;
  if (i >= 4 * 128 * 128) return;
  int c = i >> 14, r = i & 16383, k = r >> 7, n = r & 127;
  size_t o = ((size_t)(c * 128 + n)) * 128 + k;
  gwt16[o] = (_Float16)gw[i];
  wst16[o] = (_Float16)wsm[i];
}

__global__ void k_fill(float* __restrict__ p, float v, int n) {
  int i = blockIdx.x * blockDim.x + threadIdx.x;
  if (i < n) p[i] = v;
}

// deg[g][c][n] += ew over edges with dst==n  (deg pre-filled with 1.0 self loop)
__global__ void k_deg_scatter(const float* __restrict__ ea, const int* __restrict__ ei,
                              float* __restrict__ deg) {
  int idx = blockIdx.x * blockDim.x + threadIdx.x;
  if (idx >= Gn * En) return;
  int g = idx / En, e = idx % En;
  int dst = ei[g * 2 * En + En + e];
#pragma unroll
  for (int c = 0; c < 4; ++c)
    atomicAdd(&deg[(g * 4 + c) * Nn + dst], ea[(g * En + e) * 6 + 2 + c]);
}

__global__ void k_deg_fin(float* __restrict__ deg, int n) {
  int i = blockIdx.x * blockDim.x + threadIdx.x;
  if (i < n) deg[i] = rsqrtf(deg[i]);
}

// ---------------- GEMM: xw = x16 @ Wt^T  (per wave: 16x32 tile, K=128) ----------------
__global__ void k_gemm_xw(const _Float16* __restrict__ x16, const _Float16* __restrict__ wt,
                          float* __restrict__ xw) {
  const int wave = (blockIdx.x * blockDim.x + threadIdx.x) >> 5;
  const int lane = threadIdx.x & 31;
  const int g = wave >> 8, t = wave & 255;        // 64 x 4 wave-tiles per graph
  const int tm = t >> 2, tn = t & 3;
  const _Float16* A   = x16 + (size_t)g * Nn * DINn + tm * 16 * DINn;
  const _Float16* Bt0 = wt + (size_t)(tn * 32) * DINn;
  const _Float16* Bt1 = Bt0 + (size_t)16 * DINn;
  v8f c0 = {}, c1 = {};
#pragma unroll
  for (int k0 = 0; k0 < DINn; k0 += 32) {
    v16h a  = loadA(A, DINn, lane, k0);
    v16h b0 = loadBt(Bt0, DINn, lane, k0);
    v16h b1 = loadBt(Bt1, DINn, lane, k0);
    c0 = WMMA_F16(a, b0, c0);
    c1 = WMMA_F16(a, b1, c1);
  }
  const int half = lane >> 4, m = lane & 15;
  float* o = xw + ((size_t)g * Nn + tm * 16 + 8 * half) * Dn + tn * 32 + m;
#pragma unroll
  for (int r = 0; r < 8; ++r) { o[(size_t)r * Dn] = c0[r]; o[(size_t)r * Dn + 16] = c1[r]; }
}

// acc = dinv^2 * xw + b[c]   (self-loop term)
__global__ void k_acc_init(const float* __restrict__ xw, const float* __restrict__ deg,
                           const float* __restrict__ bc, float* __restrict__ acc, int c) {
  int i = blockIdx.x * blockDim.x + threadIdx.x;
  if (i >= Gn * Nn * Dn) return;
  int g = i / (Nn * Dn), r = i % (Nn * Dn), n = r / Dn, d = r % Dn;
  float dv = deg[(g * 4 + c) * Nn + n];
  acc[i] = dv * dv * xw[i] + bc[d];
}

// acc[dst] += dinv[src]*ew*dinv[dst] * xw[src]   (128 threads per edge)
__global__ void k_scatter(const float* __restrict__ ea, const int* __restrict__ ei,
                          const float* __restrict__ deg, const float* __restrict__ xw,
                          float* __restrict__ acc, int c) {
  long long tid = (long long)blockIdx.x * blockDim.x + threadIdx.x;
  if (tid >= (long long)Gn * En * 128) return;
  int d = (int)(tid & 127);
  long long eg = tid >> 7;
  int g = (int)(eg / En), e = (int)(eg % En);
  int src = ei[g * 2 * En + e];
  int dst = ei[g * 2 * En + En + e];
  float w = deg[(g * 4 + c) * Nn + src] * ea[(g * En + e) * 6 + 2 + c] *
            deg[(g * 4 + c) * Nn + dst];
  atomicAdd(&acc[((size_t)g * Nn + dst) * Dn + d], w * xw[((size_t)g * Nn + src) * Dn + d]);
}

__global__ void k_relu_h(const float* __restrict__ acc, _Float16* __restrict__ h, int n) {
  int i = blockIdx.x * blockDim.x + threadIdx.x;
  if (i < n) h[i] = (_Float16)fmaxf(acc[i], 0.0f);
}

// x_tmp += h @ Wst^T  (per wave: 16x32 tile)
__global__ void k_gemm_mix(const _Float16* __restrict__ h, const _Float16* __restrict__ wst,
                           float* __restrict__ xtmp) {
  const int wave = (blockIdx.x * blockDim.x + threadIdx.x) >> 5;
  const int lane = threadIdx.x & 31;
  const int g = wave >> 8, t = wave & 255;
  const int tm = t >> 2, tn = t & 3;
  const int half = lane >> 4, m = lane & 15;
  float* o = xtmp + ((size_t)g * Nn + tm * 16 + 8 * half) * Dn + tn * 32 + m;
  v8f c0, c1;
#pragma unroll
  for (int r = 0; r < 8; ++r) { c0[r] = o[(size_t)r * Dn]; c1[r] = o[(size_t)r * Dn + 16]; }
  const _Float16* A   = h + (size_t)g * Nn * Dn + tm * 16 * Dn;
  const _Float16* Bt0 = wst + (size_t)(tn * 32) * Dn;
  const _Float16* Bt1 = Bt0 + (size_t)16 * Dn;
#pragma unroll
  for (int k0 = 0; k0 < Dn; k0 += 32) {
    v16h a  = loadA(A, Dn, lane, k0);
    v16h b0 = loadBt(Bt0, Dn, lane, k0);
    v16h b1 = loadBt(Bt1, Dn, lane, k0);
    c0 = WMMA_F16(a, b0, c0);
    c1 = WMMA_F16(a, b1, c1);
  }
#pragma unroll
  for (int r = 0; r < 8; ++r) { o[(size_t)r * Dn] = c0[r]; o[(size_t)r * Dn + 16] = c1[r]; }
}

// gmat[b][gp*128+d] = mean_n x_tmp[g][n][d]
__global__ void k_mean(const float* __restrict__ xtmp, float* __restrict__ gmat) {
  int g = blockIdx.x, d = threadIdx.x;     // 24 blocks x 128 threads
  float s = 0.f;
  for (int n = 0; n < Nn; ++n) s += xtmp[((size_t)g * Nn + n) * Dn + d];
  gmat[(g / GPn) * (GPn * Dn) + (g % GPn) * Dn + d] = s * (1.0f / Nn);
}

// xn = f16( x_tmp / max(||row||,1e-12) ), one wave per row
__global__ void k_norm(const float* __restrict__ xtmp, _Float16* __restrict__ xn) {
  const int wave = (blockIdx.x * blockDim.x + threadIdx.x) >> 5;
  const int lane = threadIdx.x & 31;
  const float* row = xtmp + (size_t)wave * Dn;
  float s = 0.f;
#pragma unroll
  for (int q = 0; q < 4; ++q) { float v = row[lane * 4 + q]; s += v * v; }
#pragma unroll
  for (int off = 16; off > 0; off >>= 1) s += __shfl_down(s, off, 32);
  s = __shfl(s, 0, 32);
  float inv = 1.0f / fmaxf(sqrtf(s), 1e-12f);
  _Float16* orow = xn + (size_t)wave * Dn;
#pragma unroll
  for (int q = 0; q < 4; ++q) orow[lane * 4 + q] = (_Float16)(row[lane * 4 + q] * inv);
}

// all-pairs similarity: one wave per 16x64 strip of Xn_i @ Xn_j^T (16 WMMA),
// then tanh -> 21-bin -> bin-center partial sum, wave reduce, one atomic per strip.
__global__ void k_sim(const _Float16* __restrict__ xn, float* __restrict__ simacc) {
  const int wave = (blockIdx.x * blockDim.x + threadIdx.x) >> 5;
  const int lane = threadIdx.x & 31;
  const int pair = wave >> 10;            // (b,i,j); 64 x 16 wave-strips per pair
  const int t = wave & 1023;
  const int ti = t >> 4, tj = t & 15;
  const int b = pair / (GPn * GPn), ij = pair % (GPn * GPn);
  const int i = ij / GPn, j = ij % GPn;
  const _Float16* A  = xn + ((size_t)(b * GPn + i) * Nn + ti * 16) * Dn;
  const _Float16* Bj = xn + ((size_t)(b * GPn + j) * Nn + tj * 64) * Dn;
  // prefetch next strip of B rows (speculative, stays in-bounds of 6MB xn buffer)
  __builtin_prefetch(Bj + (size_t)64 * Dn, 0, 3);
  v8f c0 = {}, c1 = {}, c2 = {}, c3 = {};
#pragma unroll
  for (int k0 = 0; k0 < Dn; k0 += 32) {
    v16h a  = loadA(A, Dn, lane, k0);
    v16h b0 = loadBt(Bj, Dn, lane, k0);
    v16h b1 = loadBt(Bj + (size_t)16 * Dn, Dn, lane, k0);
    v16h b2 = loadBt(Bj + (size_t)32 * Dn, Dn, lane, k0);
    v16h b3 = loadBt(Bj + (size_t)48 * Dn, Dn, lane, k0);
    c0 = WMMA_F16(a, b0, c0);
    c1 = WMMA_F16(a, b1, c1);
    c2 = WMMA_F16(a, b2, c2);
    c3 = WMMA_F16(a, b3, c3);
  }
  float s = 0.f;
#pragma unroll
  for (int r = 0; r < 8; ++r) {
    float v0 = tanhf(c0[r]), v1 = tanhf(c1[r]), v2 = tanhf(c2[r]), v3 = tanhf(c3[r]);
    int i0 = (int)floorf((v0 + 1.0f) * 10.5f);
    int i1 = (int)floorf((v1 + 1.0f) * 10.5f);
    int i2 = (int)floorf((v2 + 1.0f) * 10.5f);
    int i3 = (int)floorf((v3 + 1.0f) * 10.5f);
    i0 = i0 < 0 ? 0 : (i0 > 20 ? 20 : i0);
    i1 = i1 < 0 ? 0 : (i1 > 20 ? 20 : i1);
    i2 = i2 < 0 ? 0 : (i2 > 20 ? 20 : i2);
    i3 = i3 < 0 ? 0 : (i3 > 20 ? 20 : i3);
    s += (i0 + i1 + i2 + i3) * 0.1f - 4.0f;      // linspace(-1,1,21)[idx] summed
  }
#pragma unroll
  for (int off = 16; off > 0; off >>= 1) s += __shfl_down(s, off, 32);
  if (lane == 0) atomicAdd(&simacc[(b * GPn + i) * GPn + j], s);
}

// tiny backnet: feat, code, strength, sim_res, r
__global__ void k_backnet(const float* __restrict__ gmat, const float* __restrict__ simacc,
                          const float* __restrict__ l1W, const float* __restrict__ l1b,
                          const float* __restrict__ cW, const float* __restrict__ cb,
                          const float* __restrict__ l2W, const float* __restrict__ l2b,
                          const float* __restrict__ sW, const float* __restrict__ sb,
                          float* __restrict__ out) {
  int b = threadIdx.x;
  if (b >= Bn) return;
  const int GD = GPn * Dn;
  float feat[20];
  for (int o = 0; o < 20; ++o) {
    float s = l1b[o];
    for (int k = 0; k < GD; ++k) s += gmat[b * GD + k] * l1W[k * 20 + o];
    feat[o] = fmaxf(s, 0.0f);
  }
  float z[2];
  for (int o = 0; o < 2; ++o) {
    float s = cb[o];
    for (int k = 0; k < 20; ++k) s += feat[k] * cW[k * 2 + o];
    z[o] = s;
  }
  float mx = fmaxf(z[0], z[1]);
  float lse = mx + logf(expf(z[0] - mx) + expf(z[1] - mx));
  float code0 = z[0] - lse, code1 = z[1] - lse;

  float sv[GPn];
  for (int i = 0; i < GPn; ++i) {
    float strength = l2b[i];
    for (int k = 0; k < GD; ++k) strength += gmat[b * GD + k] * l2W[k * GPn + i];
    float sm = 0.f;
    for (int j = 0; j < GPn; ++j) sm += simacc[(b * GPn + i) * GPn + j];
    sm *= (1.0f / ((float)Nn * (float)Nn * (float)GPn));   // mean over (n,m) and j
    sv[i] = sm * strength;
  }
  float z2[2];
  for (int o = 0; o < 2; ++o) {
    float s = sb[o];
    for (int i = 0; i < GPn; ++i) s += sv[i] * sW[i * 2 + o];
    z2[o] = s;
  }
  float mx2 = fmaxf(z2[0], z2[1]);
  float lse2 = mx2 + logf(expf(z2[0] - mx2) + expf(z2[1] - mx2));
  out[b * 2 + 0] = 0.4f * code0 + 0.6f * (z2[0] - lse2);
  out[b * 2 + 1] = 0.4f * code1 + 0.6f * (z2[1] - lse2);
  for (int o = 0; o < 20; ++o) out[Bn * 2 + b * 20 + o] = feat[o];
}

extern "C" void kernel_launch(void* const* d_in, const int* in_sizes, int n_in,
                              void* d_out, int out_size, void* d_ws, size_t ws_size,
                              hipStream_t stream) {
  const float* x     = (const float*)d_in[0];
  const float* ea    = (const float*)d_in[1];
  const float* gcnW  = (const float*)d_in[2];
  const float* gcnb  = (const float*)d_in[3];
  const float* WsM   = (const float*)d_in[4];
  const float* l1W   = (const float*)d_in[5];
  const float* l1b   = (const float*)d_in[6];
  const float* cW    = (const float*)d_in[7];
  const float* cb    = (const float*)d_in[8];
  const float* l2W   = (const float*)d_in[9];
  const float* l2b   = (const float*)d_in[10];
  const float* sW    = (const float*)d_in[11];
  const float* sb    = (const float*)d_in[12];
  const int*   ei    = (const int*)d_in[13];
  float* out = (float*)d_out;

  char* ws = (char*)d_ws;
  _Float16* x16   = (_Float16*)(ws + OFF_X16);
  _Float16* wt16  = (_Float16*)(ws + OFF_W16);    // gcn_W transposed, f16
  _Float16* wst16 = (_Float16*)(ws + OFF_WS16);   // Ws transposed, f16
  float*    deg   = (float*)(ws + OFF_DEG);
  float*    xw    = (float*)(ws + OFF_XW);
  _Float16* hbuf  = (_Float16*)(ws + OFF_XW);     // aliases xw (consumed before overwrite)
  float*    acc   = (float*)(ws + OFF_ACC);
  _Float16* xn    = (_Float16*)(ws + OFF_ACC);    // aliases acc (after channel loop)
  float*    xtmp  = (float*)(ws + OFF_XTMP);
  float*    gmat  = (float*)(ws + OFF_GMAT);
  float*    sim   = (float*)(ws + OFF_SIM);

  const int nGND = Gn * Nn * Dn;                  // 3,145,728
  // ---- conversions & init ----
  k_convert_x<<<(nGND + 255) / 256, 256, 0, stream>>>(x, x16, nGND);
  k_convert_w<<<(4 * 128 * 128 + 255) / 256, 256, 0, stream>>>(gcnW, WsM, wt16, wst16);
  k_fill<<<(Gn * 4 * Nn + 255) / 256, 256, 0, stream>>>(deg, 1.0f, Gn * 4 * Nn);
  k_fill<<<(nGND + 255) / 256, 256, 0, stream>>>(xtmp, 0.0f, nGND);
  k_fill<<<1, 256, 0, stream>>>(sim, 0.0f, Bn * GPn * GPn);
  k_deg_scatter<<<(Gn * En + 255) / 256, 256, 0, stream>>>(ea, ei, deg);
  k_deg_fin<<<(Gn * 4 * Nn + 255) / 256, 256, 0, stream>>>(deg, Gn * 4 * Nn);

  // ---- per-channel GCN + mix (WMMA GEMMs + edge scatter) ----
  for (int c = 0; c < 4; ++c) {
    const _Float16* wt16c  = wt16 + (size_t)c * 128 * 128;
    const _Float16* wst16c = wst16 + (size_t)c * 128 * 128;
    const float*    bc     = gcnb + c * 128;
    k_gemm_xw<<<768, 256, 0, stream>>>(x16, wt16c, xw);          // 6144 waves, 16x32 tiles
    k_acc_init<<<(nGND + 255) / 256, 256, 0, stream>>>(xw, deg, bc, acc, c);
    k_scatter<<<196608, 256, 0, stream>>>(ea, ei, deg, xw, acc, c);
    k_relu_h<<<(nGND + 255) / 256, 256, 0, stream>>>(acc, hbuf, nGND);
    k_gemm_mix<<<768, 256, 0, stream>>>(hbuf, wst16c, xtmp);
  }

  // ---- pooling, normalize, all-pairs similarity (dominant WMMA stage) ----
  k_mean<<<Gn, 128, 0, stream>>>(xtmp, gmat);
  k_norm<<<3072, 256, 0, stream>>>(xtmp, xn);                    // 24576 rows / 8 waves
  k_sim<<<18432, 256, 0, stream>>>(xn, sim);                     // 147456 strips (16 WMMA each)

  // ---- backnet ----
  k_backnet<<<1, 64, 0, stream>>>(gmat, sim, l1W, l1b, cW, cb, l2W, l2b, sW, sb, out);
  (void)in_sizes; (void)n_in; (void)out_size; (void)ws_size;
}
